// RNNCell_63445256896641
// MI455X (gfx1250) — matrix operand order repaired and checked by reading, not compiled, not run on hardware
//
#include <hip/hip_runtime.h>
#include <hip/hip_bf16.h>

// ---------------------------------------------------------------------------
// LSTM on MI455X (gfx1250):
//   Phase 0: convert x, W*, U* to bf16 into workspace; init h, barrier, out.
//   Phase 1: gate pre-activations via v_wmma_f32_16x16x32_bf16; each wave
//            computes a 16x64 strip (A fragment reused across 4 WMMAs).
//   Phase 2: 8 persistent workgroups; each stages 256 rows (64 hidden units x
//            4 gates) of U into 256KB LDS with the Tensor Data Mover
//            (tensor_load_to_lds + s_wait_tensorcnt), then runs the sequential
//            scan with a device-scope barrier + cluster-barrier fast path and
//            global_prefetch of the next step's pre-activations.
// ---------------------------------------------------------------------------

typedef __attribute__((ext_vector_type(16))) __bf16 v16bf;
typedef __attribute__((ext_vector_type(8)))  float  v8f;
typedef __attribute__((ext_vector_type(4)))  unsigned int v4u;
typedef __attribute__((ext_vector_type(8)))  int    v8i;
typedef __attribute__((ext_vector_type(4)))  int    v4i;

#define NXC 128
#define NHC 512
#define NGROWS 2048          // 4 gates * 512
#define NBLK 8               // persistent workgroups in phase 2
#define SLICE 64             // hidden units per workgroup

// ---- workspace layout (byte offsets, 256B aligned) ----
#define XB_OFF    0u                               // 16384 bf16      = 32768
#define WCAT_OFF  32768u                           // 2048*128 bf16   = 524288
#define UCAT_OFF  557056u                          // 2048*512 bf16   = 2097152
#define BCAT_OFF  2654208u                         // 2048 f32        = 8192
#define HBUF_OFF  2662400u                         // 512 f32         = 2048
#define CNT_OFF   2664448u                         // 1 u32 (pad 256)
#define GPRE_OFF  2664704u                         // steps*2048 f32

// ------------------------------- phase 0 -----------------------------------
__global__ void prep_x(const float* __restrict__ x, __bf16* __restrict__ xb, int n) {
    int i = blockIdx.x * blockDim.x + threadIdx.x;
    if (i < n) xb[i] = (__bf16)x[i];
}

__global__ void prep_w(const float* __restrict__ Wf, const float* __restrict__ Wi,
                       const float* __restrict__ Wo, const float* __restrict__ Wc,
                       __bf16* __restrict__ Wcat) {
    int idx = blockIdx.x * blockDim.x + threadIdx.x;      // 2048*128
    if (idx >= NGROWS * NXC) return;
    int r = idx / NXC, k = idx % NXC;
    int g = r / NHC, n = r % NHC;
    const float* W = (g == 0) ? Wf : (g == 1) ? Wi : (g == 2) ? Wo : Wc;
    Wcat[idx] = (__bf16)W[n * NXC + k];
}

__global__ void prep_u(const float* __restrict__ Uf, const float* __restrict__ Ui,
                       const float* __restrict__ Uo, const float* __restrict__ Uc,
                       __bf16* __restrict__ Ucat) {
    int idx = blockIdx.x * blockDim.x + threadIdx.x;      // 2048*512
    if (idx >= NGROWS * NHC) return;
    int r = idx / NHC, k = idx % NHC;
    int g = r / NHC, n = r % NHC;
    const float* U = (g == 0) ? Uf : (g == 1) ? Ui : (g == 2) ? Uo : Uc;
    Ucat[idx] = (__bf16)U[n * NHC + k];
}

__global__ void prep_misc(const float* __restrict__ bf_, const float* __restrict__ bi_,
                          const float* __restrict__ bo_, const float* __restrict__ bc_,
                          const float* __restrict__ by_,
                          float* __restrict__ bcat, float* __restrict__ hbuf,
                          unsigned* __restrict__ counter, float* __restrict__ out,
                          int steps) {
    int i = blockIdx.x * blockDim.x + threadIdx.x;
    if (i < NGROWS) {
        int g = i / NHC, n = i % NHC;
        const float* B = (g == 0) ? bf_ : (g == 1) ? bi_ : (g == 2) ? bo_ : bc_;
        bcat[i] = B[n];
    }
    if (i < NHC)  hbuf[i] = 0.0f;
    if (i == 0)   *counter = 0u;
    float byv = by_[0];
    for (int s = i; s < steps; s += gridDim.x * blockDim.x) out[s] = byv;
}

// ------------------------------- phase 1: WMMA GEMM ------------------------
// Gpre[s, col] = sum_{k<128} xb[s+k] * Wcat[col, k] + bcat[col]
// One wave per 16(step) x 64(col) strip; A fragment reused across 4 WMMAs.
__global__ void gates_wmma(const __bf16* __restrict__ xb,
                           const __bf16* __restrict__ Wcat,
                           const float*  __restrict__ bcat,
                           float* __restrict__ Gpre,
                           int steps) {
    const int strips_n = NGROWS / 64;                      // 32
    int wave   = (blockIdx.x * blockDim.x + threadIdx.x) >> 5;
    int lane   = threadIdx.x & 31;
    int tile_m = wave / strips_n;
    int strip  = wave % strips_n;
    if (tile_m * 16 >= steps) return;

    int half = lane >> 4;          // 0: lanes 0-15, 1: lanes 16-31
    int lrow = lane & 15;

    v8f acc0 = {}, acc1 = {}, acc2 = {}, acc3 = {};
    #pragma unroll
    for (int k0 = 0; k0 < NXC; k0 += 32) {
        // ---- A fragment: 16x32 bf16, A[M,K] = xb[(tile_m*16+M) + (k0+K)]
        v16bf a;
        int s = tile_m * 16 + lrow;
        #pragma unroll
        for (int v = 0; v < 8; ++v) {
            int kb = 2 * v + 16 * half;
            a[2 * v]     = xb[s + k0 + kb];
            a[2 * v + 1] = xb[s + k0 + kb + 1];
        }
        // ---- B fragments: B[K,N] = Wcat[strip*64 + nn*16 + N, k0+K]
        // lane: N = lrow, K = half*16 + e -> one 32B contiguous load each
        const __bf16* wbase = Wcat + (size_t)(strip * 64 + lrow) * NXC + k0 + half * 16;
        v16bf b0 = *reinterpret_cast<const v16bf*>(wbase);
        v16bf b1 = *reinterpret_cast<const v16bf*>(wbase + 16 * NXC);
        v16bf b2 = *reinterpret_cast<const v16bf*>(wbase + 32 * NXC);
        v16bf b3 = *reinterpret_cast<const v16bf*>(wbase + 48 * NXC);

        acc0 = __builtin_amdgcn_wmma_f32_16x16x32_bf16(false, a, false, b0, (short)0, acc0, false, false);
        acc1 = __builtin_amdgcn_wmma_f32_16x16x32_bf16(false, a, false, b1, (short)0, acc1, false, false);
        acc2 = __builtin_amdgcn_wmma_f32_16x16x32_bf16(false, a, false, b2, (short)0, acc2, false, false);
        acc3 = __builtin_amdgcn_wmma_f32_16x16x32_bf16(false, a, false, b3, (short)0, acc3, false, false);
    }

    // ---- D layout: VGPR v -> M = v + 8*half, N = lrow
    #pragma unroll
    for (int nn = 0; nn < 4; ++nn) {
        int col = strip * 64 + nn * 16 + lrow;
        float bias = bcat[col];
        const v8f& acc = (nn == 0) ? acc0 : (nn == 1) ? acc1 : (nn == 2) ? acc2 : acc3;
        #pragma unroll
        for (int v = 0; v < 8; ++v) {
            int m = tile_m * 16 + v + 8 * half;
            Gpre[(size_t)m * NGROWS + col] = acc[v] + bias;
        }
    }
}

// --------------------- TDM: 1-D tensor load into LDS -----------------------
// Loads `nelem` bf16 elements (contiguous) from `gptr` into LDS byte offset
// `lds_off` via the Tensor Data Mover. D# layout per CDNA5 ISA.
__device__ __forceinline__ void tdm_load_row(unsigned lds_off, const void* gptr,
                                             unsigned nelem) {
    unsigned long long ga = (unsigned long long)gptr;
    v4u g0;
    g0[0] = 1u;                                            // count=1, user mode
    g0[1] = lds_off;                                       // lds_addr (bytes)
    g0[2] = (unsigned)(ga & 0xFFFFFFFFu);                  // global_addr[31:0]
    g0[3] = (unsigned)((ga >> 32) & 0x1FFFFFFu) | (2u << 30); // addr[56:32], type=2
    v8i g1;
    g1[0] = (int)(1u << 16);                               // data_size=1 (2 bytes)
    g1[1] = (int)((nelem & 0xFFFFu) << 16);                // tensor_dim0[15:0]
    g1[2] = (int)(((nelem >> 16) & 0xFFFFu) | (1u << 16)); // dim0[31:16], tensor_dim1=1
    g1[3] = (int)((nelem & 0xFFFFu) << 16);                // tile_dim0
    g1[4] = 1;                                             // tile_dim1=1, tile_dim2=0
    g1[5] = (int)nelem;                                    // dim0_stride[31:0]
    g1[6] = (int)((nelem & 0xFFFFu) << 16);                // dim0_stride[47:32]=0, dim1_stride[15:0]
    g1[7] = (int)((nelem >> 16) & 0xFFFFu);                // dim1_stride[47:16]
    v4i z4 = {0, 0, 0, 0};
    v8i z8 = {0, 0, 0, 0, 0, 0, 0, 0};
    __builtin_amdgcn_tensor_load_to_lds(g0, g1, z4, z4, z8, 0);
}

// ------------------------------- phase 2: recurrence -----------------------
__device__ __forceinline__ float sigm(float z) { return 1.0f / (1.0f + __expf(-z)); }

__global__ void __launch_bounds__(256)
lstm_scan(const float* __restrict__ Gpre, const __bf16* __restrict__ Ucat,
          const float* __restrict__ Ahy,  float* __restrict__ out,
          float* __restrict__ hbuf, unsigned* __restrict__ counter, int steps) {
    extern __shared__ unsigned char smem[];
    __bf16* Ulds = (__bf16*)smem;                          // 256 rows * 512 * 2B = 256KB
    float*  hsh  = (float*)(smem + 256u * 512u * 2u);      // 512 f32
    float*  dots = hsh + NHC;                              // 256 f32

    const int b   = blockIdx.x;          // 0..7
    const int tid = threadIdx.x;         // 0..255

    // ---- stage this WG's 4 x (64 rows x 512) U blocks into LDS via TDM ----
    if (tid < 32) {                       // wave 0 only; TDM ignores EXEC
        #pragma unroll
        for (int gg = 0; gg < 4; ++gg) {
            const void* src = (const void*)(Ucat + ((size_t)gg * NHC + (size_t)b * SLICE) * NHC);
            tdm_load_row((unsigned)(gg * SLICE * NHC * 2), src, SLICE * NHC);
        }
        __builtin_amdgcn_s_wait_tensorcnt(0);
    }
    __syncthreads();

    float c_reg = 0.0f;

    for (int s = 0; s < steps; ++s) {
        // stage h (released by all WGs at end of previous step) into LDS
        { float2 hv = ((const float2*)hbuf)[tid];
          ((float2*)hsh)[tid] = hv; }
        // prefetch next step's pre-activation chunks for this WG
        if (tid < 4 && s + 1 < steps)
            __builtin_prefetch(Gpre + (size_t)(s + 1) * NGROWS + tid * NHC + b * SLICE, 0, 1);
        __syncthreads();

        // 256 dot products of length 512 out of LDS
        float acc = 0.0f;
        const uint4* urow4 = (const uint4*)(Ulds + (size_t)tid * NHC);
        #pragma unroll 4
        for (int k8 = 0; k8 < NHC / 8; ++k8) {
            uint4 u = urow4[k8];
            const __bf16* up = (const __bf16*)&u;
            const float*  hp = hsh + k8 * 8;
            #pragma unroll
            for (int e = 0; e < 8; ++e) acc += (float)up[e] * hp[e];
        }
        dots[tid] = acc;
        __syncthreads();

        if (tid < SLICE) {
            int jj = b * SLICE + tid;
            const float* gp = Gpre + (size_t)s * NGROWS;
            float zf = gp[0 * NHC + jj] + dots[0 * SLICE + tid];
            float zi = gp[1 * NHC + jj] + dots[1 * SLICE + tid];
            float zo = gp[2 * NHC + jj] + dots[2 * SLICE + tid];
            float zc = gp[3 * NHC + jj] + dots[3 * SLICE + tid];
            float f  = sigm(zf), ii = sigm(zi), o = sigm(zo), gv = tanhf(zc);
            c_reg = f * c_reg + ii * gv;
            float h = o * tanhf(c_reg);
            hbuf[jj] = h;
            // partial mu contribution, wave-reduced (wave32) then one atomic
            float pm = Ahy[jj] * h;
            #pragma unroll
            for (int off = 16; off; off >>= 1) pm += __shfl_down(pm, off, 32);
            if ((tid & 31) == 0) atomicAdd(&out[s], pm);
        }
        __threadfence();
        __syncthreads();

        // inter-workgroup barrier: cluster fast path (NOP if not clustered)
        // + device-scope atomic generation barrier for safety.
        if (tid == 0) {
            __builtin_amdgcn_s_cluster_barrier();
            __hip_atomic_fetch_add(counter, 1u, __ATOMIC_ACQ_REL, __HIP_MEMORY_SCOPE_AGENT);
            unsigned target = (unsigned)(s + 1) * NBLK;
            while (__hip_atomic_load(counter, __ATOMIC_ACQUIRE, __HIP_MEMORY_SCOPE_AGENT) < target)
                __builtin_amdgcn_s_sleep(2);
        }
        __syncthreads();
    }
}

// ------------------------------- launcher ----------------------------------
extern "C" void kernel_launch(void* const* d_in, const int* in_sizes, int n_in,
                              void* d_out, int out_size, void* d_ws, size_t ws_size,
                              hipStream_t stream) {
    const float* x_t = (const float*)d_in[0];
    const float* Wf  = (const float*)d_in[1];
    const float* Uf  = (const float*)d_in[2];
    const float* bf_ = (const float*)d_in[3];
    const float* Wi  = (const float*)d_in[4];
    const float* Ui  = (const float*)d_in[5];
    const float* bi_ = (const float*)d_in[6];
    const float* Wo  = (const float*)d_in[7];
    const float* Uo  = (const float*)d_in[8];
    const float* bo_ = (const float*)d_in[9];
    const float* Wc  = (const float*)d_in[10];
    const float* Uc  = (const float*)d_in[11];
    const float* bc_ = (const float*)d_in[12];
    const float* Ahy = (const float*)d_in[13];
    const float* by_ = (const float*)d_in[14];
    float* out = (float*)d_out;

    const int T     = in_sizes[0];
    const int steps = T - NXC;            // 16256

    char* ws = (char*)d_ws;
    __bf16*   xb      = (__bf16*)(ws + XB_OFF);
    __bf16*   Wcat    = (__bf16*)(ws + WCAT_OFF);
    __bf16*   Ucat    = (__bf16*)(ws + UCAT_OFF);
    float*    bcat    = (float*)(ws + BCAT_OFF);
    float*    hbuf    = (float*)(ws + HBUF_OFF);
    unsigned* counter = (unsigned*)(ws + CNT_OFF);
    float*    Gpre    = (float*)(ws + GPRE_OFF);

    // phase 0
    prep_x<<<(T + 255) / 256, 256, 0, stream>>>(x_t, xb, T);
    prep_w<<<(NGROWS * NXC + 255) / 256, 256, 0, stream>>>(Wf, Wi, Wo, Wc, Wcat);
    prep_u<<<(NGROWS * NHC + 255) / 256, 256, 0, stream>>>(Uf, Ui, Uo, Uc, Ucat);
    prep_misc<<<(steps + 255) / 256, 256, 0, stream>>>(bf_, bi_, bo_, bc_, by_,
                                                       bcat, hbuf, counter, out, steps);

    // phase 1: one wave per 16x64 strip; 8 waves per block
    int waves = (steps / 16) * (NGROWS / 64);              // 1016 * 32
    gates_wmma<<<waves / 8, 256, 0, stream>>>(xb, Wcat, bcat, Gpre, steps);

    // phase 2: 8 persistent workgroups, 256KB+3KB dynamic LDS each
    size_t smem = 256u * 512u * 2u + (NHC + 256) * sizeof(float);
    lstm_scan<<<NBLK, 256, smem, stream>>>(Gpre, Ucat, Ahy, out, hbuf, counter, steps);
}
